// PointerAC_39195871543573
// MI455X (gfx1250) — matrix operand order repaired and checked by reading, not compiled, not run on hardware
//
#include <hip/hip_runtime.h>

// ---------------------------------------------------------------------------
// PointerAC on MI455X (gfx1250, wave32).
//
// Roofline: the only dense work is step-1 scores = X_onehot @ W_in
// (M=64, K=N=7680). X is block-one-hot (30 ones per row), so only the
// K-blocks covering each sample's assembly matter (<=2 of 240 per sample).
// We run it through v_wmma_f32_16x16x32_f16 (0/1 values are exact in f16,
// f32 accumulate of <=30 ones is exact) with wave-uniform K-block skipping,
// so traffic stays ~60MB (~2.5us at 23.3 TB/s). Everything else (column
// sums, 8 sparse recurrent steps, kWTA, argmax) is tiny and done with
// LDS-resident per-sample blocks.
// ---------------------------------------------------------------------------

#define NN     256
#define CAPW   30
#define NNEUR  7680     // NN * CAPW
#define BATCH  64
#define KMAX   8

typedef __attribute__((ext_vector_type(16))) _Float16 v16h;
typedef __attribute__((ext_vector_type(8)))  float    v8f;

// ---------------------------------------------------------------------------
// Kernel 1: den[j] = max(col_sum(W_enc)[j], 1e-6)
// col_sum[j] = sum over assemblies a with p[a]==node_of(j), rows i in a,
// of W_rec[i][j] -- accumulated in ascending-i order to match the reference.
// ---------------------------------------------------------------------------
__global__ __launch_bounds__(256) void k_den(const int* __restrict__ p,
                                             const float* __restrict__ Wrec,
                                             float* __restrict__ den) {
    int j = blockIdx.x * blockDim.x + threadIdx.x;
    if (j >= NNEUR) return;
    int c = j / CAPW;
    float s = 0.0f;
    for (int a = 0; a < NN; ++a) {
        if (p[a] == c) {
            const float* row = Wrec + (size_t)a * CAPW * NNEUR + j;
            for (int r = 0; r < CAPW; ++r) s += row[(size_t)r * NNEUR];
        }
    }
    den[j] = fmaxf(s, 1e-6f);
}

// ---------------------------------------------------------------------------
// Kernel 2: step-1 scores via WMMA.  One wave (32 threads) per 16x16 output
// tile.  A[m][k] = 1 iff k in [s[m]*30, s[m]*30+30), generated in registers.
// B tile = W_in[k, n] converted f32->f16 (exact: values are 0/1).
// K-blocks with an all-zero A tile are skipped wave-uniformly via __ballot.
//
// 16-bit A 16x32 VGPR layout (ISA 7.12.2): lanes 0-15 are M=lane holding
// K = {0..7, 16..23}; lanes 16-31 are M=lane-16 holding K = {8..15, 24..31}.
// B 32x16: lanes 0-15 hold K = {0..7, 16..23} for N=lane, lanes 16-31 hold K = {8..15, 24..31}.
// C/D f32: lanes 0-15 => N=lane, VGPR r = row m0+r; lanes 16-31 => rows m0+8+r.
// ---------------------------------------------------------------------------
__global__ __launch_bounds__(32) void k_step1_wmma(const int* __restrict__ s,
                                                   const float* __restrict__ Win,
                                                   float* __restrict__ scores) {
    const int l  = threadIdx.x;              // lane 0..31 (wave32)
    const int n0 = blockIdx.x * 16;
    const int m0 = blockIdx.y * 16;
    const bool hiHalf = (l >= 16);
    const int  lm = l & 15;

    const int asmb = s[m0 + lm];             // assembly for this lane's M row
    const int lo = asmb * CAPW;
    const int hi = lo + CAPW;
    const int n  = n0 + lm;

    v8f acc = {};
    for (int kb = 0; kb < NNEUR / 32; ++kb) {
        const int kbase = kb * 32;
        // Does any of the 16 sample assemblies intersect this K block?
        bool need = (lo < kbase + 32) && (hi > kbase);
        if (__ballot(need) == 0ull) continue;          // wave-uniform skip

        // Build A tile analytically (one-hot block indicator).
        v16h a;
        #pragma unroll
        for (int e = 0; e < 16; ++e) {
            int koff = hiHalf ? (e < 8 ? e + 8 : e + 16)
                              : (e < 8 ? e     : e + 8);
            int kk = kbase + koff;
            a[e] = (kk >= lo && kk < hi) ? (_Float16)1.0f : (_Float16)0.0f;
        }
        // Load B tile (32 rows of W_in x 16 cols), f32 -> f16 (exact 0/1).
        v16h b;
        const int kB = kbase + (hiHalf ? 8 : 0);
        #pragma unroll
        for (int e = 0; e < 16; ++e)
            b[e] = (_Float16)Win[(size_t)(kB + (e < 8 ? e : e + 8)) * NNEUR + n];

        acc = __builtin_amdgcn_wmma_f32_16x16x32_f16(
            /*neg_a=*/false, a, /*neg_b=*/false, b,
            /*c_mod=*/(short)0, acc, /*reuse_a=*/false, /*reuse_b=*/false);
    }

    const int mb = m0 + (hiHalf ? 8 : 0);
    #pragma unroll
    for (int r = 0; r < 8; ++r)
        scores[(size_t)(mb + r) * NNEUR + n] = acc[r];
}

// ---------------------------------------------------------------------------
// kWTA on an LDS-resident score vector: extract top-30 by 30 sequential
// argmax reductions (min-index tie-break == lax.top_k stable order),
// then sort the selected indices ascending (to match the reference's
// ascending-i accumulation order in the next step).
// ---------------------------------------------------------------------------
__device__ void kwta_select(float* sc, int* sel, float* rv, int* ri) {
    const int tid = threadIdx.x;
    for (int it = 0; it < CAPW; ++it) {
        float bv = -1.0f; int bi = NNEUR;     // scores are >= 0
        for (int j = tid; j < NNEUR; j += 256) {
            float v = sc[j];
            if (v > bv) { bv = v; bi = j; }   // ascending j => first occurrence
        }
        rv[tid] = bv; ri[tid] = bi;
        __syncthreads();
        for (int off = 128; off > 0; off >>= 1) {
            if (tid < off) {
                float ov = rv[tid + off]; int oi = ri[tid + off];
                if (ov > rv[tid] || (ov == rv[tid] && oi < ri[tid])) {
                    rv[tid] = ov; ri[tid] = oi;
                }
            }
            __syncthreads();
        }
        if (tid == 0) { sel[it] = ri[0]; sc[ri[0]] = -3.0e38f; }
        __syncthreads();
    }
    if (tid == 0) {                           // insertion sort, 30 elements
        for (int i = 1; i < CAPW; ++i) {
            int key = sel[i]; int j = i - 1;
            while (j >= 0 && sel[j] > key) { sel[j + 1] = sel[j]; --j; }
            sel[j + 1] = key;
        }
    }
    __syncthreads();
}

// ---------------------------------------------------------------------------
// Kernel 3: kWTA of step-1 scores -> initial active list act[b][0..29].
// ---------------------------------------------------------------------------
__global__ __launch_bounds__(256) void k_kwta1(const float* __restrict__ scores,
                                               int* __restrict__ act) {
    __shared__ float sc[NNEUR];
    __shared__ float rv[256];
    __shared__ int   ri[256];
    __shared__ int   sel[CAPW];
    const int b = blockIdx.x, tid = threadIdx.x;
    for (int j = tid; j < NNEUR; j += 256)
        sc[j] = scores[(size_t)b * NNEUR + j];
    __syncthreads();
    kwta_select(sc, sel, rv, ri);
    if (tid < CAPW) act[b * CAPW + tid] = sel[tid];
}

// ---------------------------------------------------------------------------
// Kernel 4: one recurrent pointer-chase step (masked by t < k[b]).
// score[j] = sum over active i (ascending) of W_rec[i][j]/den[j] when
// node_of(j) == p[node_of(i)]; then kWTA.
// ---------------------------------------------------------------------------
__global__ __launch_bounds__(256) void k_step(const int* __restrict__ p,
                                              const int* __restrict__ kv,
                                              const float* __restrict__ Wrec,
                                              const float* __restrict__ den,
                                              int* __restrict__ act, int t) {
    const int b = blockIdx.x, tid = threadIdx.x;
    if (t >= kv[b]) return;                   // keep previous x (jnp.where)
    __shared__ float sc[NNEUR];
    __shared__ float rv[256];
    __shared__ int   ri[256];
    __shared__ int   sel[CAPW];
    __shared__ int   ai[CAPW];
    __shared__ int   tc[CAPW];
    if (tid < CAPW) {
        int i = act[b * CAPW + tid];
        ai[tid] = i;
        tc[tid] = p[i / CAPW];                // target assembly of source i
    }
    __syncthreads();
    for (int j = tid; j < NNEUR; j += 256) {
        const int   c = j / CAPW;
        const float d = den[j];
        float sum = 0.0f;
        for (int ii = 0; ii < CAPW; ++ii)     // ai sorted ascending
            if (tc[ii] == c)
                sum += Wrec[(size_t)ai[ii] * NNEUR + j] / d;
        sc[j] = sum;
    }
    __syncthreads();
    kwta_select(sc, sel, rv, ri);
    if (tid < CAPW) act[b * CAPW + tid] = sel[tid];
}

// ---------------------------------------------------------------------------
// Kernel 5: overlaps -> argmax (lowest index on ties) -> one-hot logits.
// ---------------------------------------------------------------------------
__global__ __launch_bounds__(256) void k_final(const int* __restrict__ act,
                                               float* __restrict__ out) {
    __shared__ int cnt[NN];
    __shared__ int best;
    const int b = blockIdx.x, tid = threadIdx.x;
    cnt[tid] = 0;
    __syncthreads();
    if (tid < CAPW) atomicAdd(&cnt[act[b * CAPW + tid] / CAPW], 1);
    __syncthreads();
    if (tid == 0) {
        int bb = 0, bc = cnt[0];
        for (int c = 1; c < NN; ++c)
            if (cnt[c] > bc) { bc = cnt[c]; bb = c; }
        best = bb;
    }
    __syncthreads();
    out[b * NN + tid] = (tid == best) ? 1.0f : 0.0f;
}

// ---------------------------------------------------------------------------
// Host-side launcher. ws layout: den[7680] f32 | scores[64*7680] f32 |
// act[64*30] int  (~2.0 MB total).
// ---------------------------------------------------------------------------
extern "C" void kernel_launch(void* const* d_in, const int* in_sizes, int n_in,
                              void* d_out, int out_size, void* d_ws, size_t ws_size,
                              hipStream_t stream) {
    const int*   p    = (const int*)d_in[0];
    const int*   s    = (const int*)d_in[1];
    const int*   kv   = (const int*)d_in[2];
    const float* Win  = (const float*)d_in[3];
    const float* Wrec = (const float*)d_in[4];
    float*       out  = (float*)d_out;

    float* den    = (float*)d_ws;
    float* scores = den + NNEUR;
    int*   act    = (int*)(scores + (size_t)BATCH * NNEUR);

    k_den<<<(NNEUR + 255) / 256, 256, 0, stream>>>(p, Wrec, den);
    k_step1_wmma<<<dim3(NNEUR / 16, BATCH / 16), 32, 0, stream>>>(s, Win, scores);
    k_kwta1<<<BATCH, 256, 0, stream>>>(scores, act);
    for (int t = 0; t < KMAX; ++t)
        k_step<<<BATCH, 256, 0, stream>>>(p, kv, Wrec, den, act, t);
    k_final<<<BATCH, 256, 0, stream>>>(act, out);
}
